// RCNN_13434657702227
// MI455X (gfx1250) — matrix operand order repaired and checked by reading, not compiled, not run on hardware
//
#include <hip/hip_runtime.h>
#include <hip/hip_bf16.h>
#include <math.h>

// ---------------------------------------------------------------------------
// RCNN forward for MI455X (gfx1250): all GEMMs via v_wmma_f32_16x16x32_f16.
// Projection outputs are stored in WMMA C-fragment layout (time-major tiles)
// so the sequential scan initializes accumulators with one b128 load/lane.
// ---------------------------------------------------------------------------

typedef __attribute__((ext_vector_type(16))) _Float16 v16h;
typedef __attribute__((ext_vector_type(8)))  _Float16 v8h;
typedef __attribute__((ext_vector_type(8)))  float    v8f;

#define NEMB 50000
#define EDIM 300
#define EPAD 320      // 300 padded to 10 * 32
#define CSZ  256
#define HSZ  512
#define BB   64
#define LL   512
#define ROWS (BB * LL)            // 32768
#define KCAT 832                  // 256 (cl) + 320 (emb pad) + 256 (cr) = 26*32
#define COL_EMB 256
#define COL_CR  576
#define AS_STRIDE 264             // 256 + 8 halves LDS pad (bank spread)

static __device__ __forceinline__ v8f wmma_f16(v16h a, v16h b, v8f c) {
    return __builtin_amdgcn_wmma_f32_16x16x32_f16(
        /*neg_a=*/false, a, /*neg_b=*/false, b,
        /*c_mod=*/(short)0, c, /*reuse_a=*/false, /*reuse_b=*/false);
}

// A fragment (16x32 f16) from per-lane row pointer. ISA layout: lane m holds
// halves [0..7] = K kbase..kbase+7, [8..15] = K kbase+16..kbase+23 with
// kbase = kc*32 + ((lane>>4)<<3).
static __device__ __forceinline__ v16h load_a_frag_ptr(const _Float16* __restrict__ rowp,
                                                       int lane, int kc) {
    int kbase = kc * 32 + ((lane >> 4) << 3);
    v8h lo = *(const v8h*)(rowp + kbase);
    v8h hi = *(const v8h*)(rowp + kbase + 16);
    v16h a;
#pragma unroll
    for (int i = 0; i < 8; ++i) { a[i] = lo[i]; a[i + 8] = hi[i]; }
    return a;
}

static __device__ __forceinline__ v16h load_a_frag(const _Float16* __restrict__ base,
                                                   int lane, int kc, int ld) {
    return load_a_frag_ptr(base + (size_t)(lane & 15) * ld, lane, kc);
}

// B fragment (32x16 f16): lane holds column N = n0 + (lane & 15); halves
// [0..7] = K kbase..kbase+7, [8..15] = K kbase+16..kbase+23 (same as A).
static __device__ __forceinline__ v16h load_b_frag(const _Float16* __restrict__ B,
                                                   int lane, int kc, int ldb, int n0) {
    int kbase = kc * 32 + ((lane >> 4) << 3);
    const _Float16* colp = B + n0 + (lane & 15);
    v16h b;
#pragma unroll
    for (int i = 0; i < 8; ++i) {
        b[i] = colp[(size_t)(kbase + i) * ldb];
        b[i + 8] = colp[(size_t)(kbase + 16 + i) * ldb];
    }
    return b;
}

// --------------------------- prep kernels ----------------------------------

__global__ void k_cvt_pad(const float* __restrict__ src, _Float16* __restrict__ dst,
                          int Ks, int Kp, int N) {
    int idx = blockIdx.x * 256 + threadIdx.x;
    if (idx >= Kp * N) return;
    int k = idx / N, n = idx % N;
    dst[idx] = (k < Ks) ? (_Float16)src[(size_t)k * N + n] : (_Float16)0.f;
}

// W2 rows are x-features [cl 0..255 | emb 256..555 | cr 556..811]; our xcat has
// a 20-col zero pad between emb and cr, so remap rows with a gap at 556..575.
__global__ void k_cvt_w2(const float* __restrict__ src, _Float16* __restrict__ dst) {
    int idx = blockIdx.x * 256 + threadIdx.x;
    if (idx >= KCAT * HSZ) return;
    int k = idx / HSZ, n = idx % HSZ;
    float v = 0.f;
    if (k < 556)       v = src[(size_t)k * HSZ + n];
    else if (k >= 576) v = src[(size_t)(k - 20) * HSZ + n];
    dst[idx] = (_Float16)v;
}

__global__ __launch_bounds__(320) void k_gather(const int* __restrict__ seq_ids,
                                                const float* __restrict__ emb,
                                                _Float16* __restrict__ xcat) {
    int row = blockIdx.x;            // b*L + l
    int e = threadIdx.x;             // 0..319
    int id = seq_ids[row];
    float v = (e < EDIM) ? emb[(size_t)id * EDIM + e] : 0.f;
    xcat[(size_t)row * KCAT + COL_EMB + e] = (_Float16)v;
}

__global__ void k_init_y3(unsigned* __restrict__ y3key) {
    int i = blockIdx.x * 256 + threadIdx.x;
    if (i < BB * HSZ) y3key[i] = 0u;
}

// ------------------------ projection GEMM ----------------------------------
// Time-major M remap: row' = l*64 + b, so a 16-row tile = 16 batch rows at one
// timestep (= one scan tile). Output stored in C-fragment layout: per tile,
// lane stores its 8 accs as one contiguous v8h at tile*256 + lane*8.
__global__ __launch_bounds__(256) void k_proj(const _Float16* __restrict__ xcat,
                                              const _Float16* __restrict__ W,
                                              _Float16* __restrict__ outfrag) {
    int wave = threadIdx.x >> 5, lane = threadIdx.x & 31;
    int tile  = blockIdx.x * 8 + wave;      // 32768 tiles
    int mtile = tile >> 4;                  // = l*4 + bslice
    int n0    = (tile & 15) * 16;
    int l = mtile >> 2;
    int b = ((mtile & 3) << 4) + (lane & 15);
    const _Float16* arow = xcat + ((size_t)b * LL + l) * KCAT + COL_EMB;
    v8f acc0 = {}, acc1 = {};
#pragma unroll
    for (int kc = 0; kc < EPAD / 32; kc += 2) {
        __builtin_prefetch(W + (size_t)((kc + 2) * 32 + lane) * CSZ + n0, 0, 0);
        acc0 = wmma_f16(load_a_frag_ptr(arow, lane, kc),
                        load_b_frag(W, lane, kc, CSZ, n0), acc0);
        acc1 = wmma_f16(load_a_frag_ptr(arow, lane, kc + 1),
                        load_b_frag(W, lane, kc + 1, CSZ, n0), acc1);
    }
    v8h o;
#pragma unroll
    for (int i = 0; i < 8; ++i) o[i] = (_Float16)(acc0[i] + acc1[i]);
    *(v8h*)(outfrag + (size_t)tile * 256 + lane * 8) = o;
}

// ----------------------------- the scan ------------------------------------
// 8 workgroups: blockIdx = {batch-slice 0..3} x {dir 0..1}. W (256x256 f16)
// lives in VGPRs as B fragments for all 512 steps; 16x256 state lives in LDS.
// x_t fragments arrive as one software-pipelined b128 load per lane.
__global__ __launch_bounds__(512) void k_scan(const _Float16* __restrict__ Wl16,
                                              const _Float16* __restrict__ Wr16,
                                              const _Float16* __restrict__ xl16,
                                              const _Float16* __restrict__ xr16,
                                              _Float16* __restrict__ xcat) {
    int dir = blockIdx.x & 1;
    int ms  = blockIdx.x >> 1;
    int b0  = ms * 16;
    int wave = threadIdx.x >> 5, lane = threadIdx.x & 31;
    int n0 = wave * 16;                         // this wave's 16 output columns
    const _Float16* W   = dir ? Wr16 : Wl16;
    const _Float16* xin = dir ? xr16 : xl16;
    int colbase = dir ? COL_CR : 0;

    __shared__ _Float16 As[16][AS_STRIDE];
    for (int i = threadIdx.x; i < 16 * AS_STRIDE; i += 512)
        (&As[0][0])[i] = (_Float16)0.f;         // c0 = 0

    v16h Bf[8];                                 // resident weight fragments
#pragma unroll
    for (int kc = 0; kc < 8; ++kc)
        Bf[kc] = load_b_frag(W, lane, kc, CSZ, n0);
    __syncthreads();

    const int c = lane & 15;
    const int rsel = (lane < 16) ? 0 : 8;

    // fragment-layout x: tile index for (time tt, slice ms) = (tt*4+ms)*16+wave
    auto xfrag_ptr = [&](int t) {
        int tt = dir ? (LL - 1 - t) : t;
        return (const v8h*)(xin + ((size_t)((tt * 4 + ms) * 16 + wave)) * 256 + lane * 8);
    };

    v8h xcur = *xfrag_ptr(0);
    for (int t = 0; t < LL; ++t) {
        v8h xnxt;
        if (t + 1 < LL) {
            __builtin_prefetch((const void*)xfrag_ptr(t + 1 < LL - 1 ? t + 2 : t + 1), 0, 0);
            xnxt = *xfrag_ptr(t + 1);           // off-chain: independent of state
        }
        v8f acc0, acc1 = {};
#pragma unroll
        for (int i = 0; i < 8; ++i) acc0[i] = (float)xcur[i];
#pragma unroll
        for (int kc = 0; kc < 8; kc += 2) {     // two independent WMMA chains
            acc0 = wmma_f16(load_a_frag(&As[0][0], lane, kc,     AS_STRIDE), Bf[kc],     acc0);
            acc1 = wmma_f16(load_a_frag(&As[0][0], lane, kc + 1, AS_STRIDE), Bf[kc + 1], acc1);
        }
        __syncthreads();                        // all waves done reading As
#pragma unroll
        for (int i = 0; i < 8; ++i) {
            int r = rsel + i;
            _Float16 h = (_Float16)tanhf(acc0[i] + acc1[i]);
            As[r][n0 + c] = h;                  // state for step t+1
            // fire-and-forget global store (never waited on inside the loop)
            xcat[((size_t)(b0 + r) * LL + t) * KCAT + colbase + n0 + c] = h;
        }
        __syncthreads();                        // state visible to all waves
        xcur = xnxt;
    }
}

// ----------------------- hidden GEMM + max fuse ----------------------------
// y2 = tanh(xcat @ W2 + b2); y3 = max over l. A 16-row tile is one batch idx,
// so reduce tile-column maxes and atomicMax into monotone-uint keys.
__global__ __launch_bounds__(256) void k_hidden(const _Float16* __restrict__ xcat,
                                                const _Float16* __restrict__ W2h,
                                                const float* __restrict__ b2,
                                                unsigned* __restrict__ y3key) {
    int wave = threadIdx.x >> 5, lane = threadIdx.x & 31;
    int tile  = blockIdx.x * 8 + wave;          // 65536 tiles
    int mtile = tile >> 5;
    int n0    = (tile & 31) * 16;
    const _Float16* atile = xcat + (size_t)(mtile * 16) * KCAT;
    v8f acc0 = {}, acc1 = {};
#pragma unroll 2
    for (int kc = 0; kc < KCAT / 32; kc += 2) {
        __builtin_prefetch(atile + (size_t)(lane & 15) * KCAT + (kc + 2) * 32, 0, 0);
        acc0 = wmma_f16(load_a_frag(atile, lane, kc, KCAT),
                        load_b_frag(W2h, lane, kc, HSZ, n0), acc0);
        acc1 = wmma_f16(load_a_frag(atile, lane, kc + 1, KCAT),
                        load_b_frag(W2h, lane, kc + 1, HSZ, n0), acc1);
    }
    float bn = b2[n0 + (lane & 15)];
    float m = -1e30f;
#pragma unroll
    for (int i = 0; i < 8; ++i)
        m = fmaxf(m, tanhf(acc0[i] + acc1[i] + bn));
    m = fmaxf(m, __shfl_xor(m, 16));            // combine row halves
    if (lane < 16) {
        int b = mtile >> 5;                     // 32 mtiles per batch index
        unsigned u = __float_as_uint(m);
        unsigned key = (u & 0x80000000u) ? ~u : (u | 0x80000000u);
        atomicMax(&y3key[(size_t)b * HSZ + n0 + lane], key);
    }
}

// --------------------------- final GEMV ------------------------------------
__global__ __launch_bounds__(256) void k_final(const unsigned* __restrict__ y3key,
                                               const float* __restrict__ W_out,
                                               const float* __restrict__ b_out,
                                               float* __restrict__ out) {
    int b = blockIdx.x, tid = threadIdx.x;
    float s = 0.f;
    for (int h = tid; h < HSZ; h += 256) {
        unsigned key = y3key[(size_t)b * HSZ + h];
        unsigned u = (key & 0x80000000u) ? (key ^ 0x80000000u) : ~key;
        s += __uint_as_float(u) * W_out[h];
    }
#pragma unroll
    for (int off = 16; off; off >>= 1) s += __shfl_down(s, off);
    __shared__ float red[8];
    if ((tid & 31) == 0) red[tid >> 5] = s;
    __syncthreads();
    if (tid == 0) {
        float t = 0.f;
        for (int i = 0; i < 8; ++i) t += red[i];
        out[b] = t + b_out[0];
    }
}

// --------------------------- host driver -----------------------------------
extern "C" void kernel_launch(void* const* d_in, const int* in_sizes, int n_in,
                              void* d_out, int out_size, void* d_ws, size_t ws_size,
                              hipStream_t stream) {
    const int*   seq_ids = (const int*)  d_in[0];
    const float* emb     = (const float*)d_in[1];
    const float* W_l     = (const float*)d_in[2];
    const float* W_r     = (const float*)d_in[3];
    const float* W_sl    = (const float*)d_in[4];
    const float* W_sr    = (const float*)d_in[5];
    const float* W2      = (const float*)d_in[6];
    const float* b2      = (const float*)d_in[7];
    const float* W_out   = (const float*)d_in[8];
    const float* b_out   = (const float*)d_in[9];
    float* out = (float*)d_out;

    size_t off = 0;
    auto carve = [&](size_t bytes) {
        void* p = (char*)d_ws + off;
        off += (bytes + 255) & ~(size_t)255;
        return p;
    };
    _Float16* XCAT  = (_Float16*)carve((size_t)ROWS * KCAT * 2);   // 54.5 MB
    _Float16* XL    = (_Float16*)carve((size_t)ROWS * CSZ * 2);    // 16.8 MB (frag layout)
    _Float16* XR    = (_Float16*)carve((size_t)ROWS * CSZ * 2);    // 16.8 MB (frag layout)
    _Float16* WL16  = (_Float16*)carve((size_t)CSZ * CSZ * 2);
    _Float16* WR16  = (_Float16*)carve((size_t)CSZ * CSZ * 2);
    _Float16* WSL16 = (_Float16*)carve((size_t)EPAD * CSZ * 2);
    _Float16* WSR16 = (_Float16*)carve((size_t)EPAD * CSZ * 2);
    _Float16* W216  = (_Float16*)carve((size_t)KCAT * HSZ * 2);
    unsigned* Y3KEY = (unsigned*)carve((size_t)BB * HSZ * 4);
    (void)ws_size; (void)in_sizes; (void)n_in; (void)out_size;

    // weight conversion (fp32 -> f16, K-padded)
    k_cvt_pad<<<(CSZ * CSZ + 255) / 256, 256, 0, stream>>>(W_l, WL16, CSZ, CSZ, CSZ);
    k_cvt_pad<<<(CSZ * CSZ + 255) / 256, 256, 0, stream>>>(W_r, WR16, CSZ, CSZ, CSZ);
    k_cvt_pad<<<(EPAD * CSZ + 255) / 256, 256, 0, stream>>>(W_sl, WSL16, EDIM, EPAD, CSZ);
    k_cvt_pad<<<(EPAD * CSZ + 255) / 256, 256, 0, stream>>>(W_sr, WSR16, EDIM, EPAD, CSZ);
    k_cvt_w2<<<(KCAT * HSZ + 255) / 256, 256, 0, stream>>>(W2, W216);
    k_init_y3<<<(BB * HSZ + 255) / 256, 256, 0, stream>>>(Y3KEY);

    // embedding gather straight into the concat buffer (middle columns)
    k_gather<<<ROWS, 320, 0, stream>>>(seq_ids, emb, XCAT);

    // input projections (WMMA, fragment-layout outputs)
    k_proj<<<4096, 256, 0, stream>>>(XCAT, WSL16, XL);
    k_proj<<<4096, 256, 0, stream>>>(XCAT, WSR16, XR);

    // bidirectional recurrence: 8 independent chains (4 batch slices x 2 dirs)
    k_scan<<<8, 512, 0, stream>>>(WL16, WR16, XL, XR, XCAT);

    // hidden GEMM + tanh + max-over-L (fused)
    k_hidden<<<8192, 256, 0, stream>>>(XCAT, W216, b2, Y3KEY);

    // output GEMV
    k_final<<<BB, 256, 0, stream>>>(Y3KEY, W_out, b_out, out);
}
